// LocalitySensitiveHashing_29111288333011
// MI455X (gfx1250) — matrix operand order repaired and checked by reading, not compiled, not run on hardware
//
#include <hip/hip_runtime.h>

// CDNA5 / gfx1250 LSH kernel.
// NOTE: the reference output is provably all-zeros (|sim| <= 1/2048 < 0.3), so
// the roofline optimum is a 33.5 MB zero-store (~1.4 us @ 23.3 TB/s). We ship
// the faithful fused pipeline: bf16 WMMA is provably safe here (sign flips
// near zero cannot cross the 0.3 threshold).
//
// This round: peel the topology term (K=256) into its own 8-step loop so the
// main 24-step K-loop has zero scalar control flow -- just 2x global b128 (A),
// 2x ds b128 (B), 1x v_wmma per step.

typedef __attribute__((ext_vector_type(16))) __bf16 v16bf;
typedef __attribute__((ext_vector_type(8)))  __bf16 v8bf;
typedef __attribute__((ext_vector_type(4)))  __bf16 v4bf;
typedef __attribute__((ext_vector_type(8)))  float  v8f;

#define B_  2
#define SQ_ 2048
#define SK_ 2048
#define D_  1024
#define H_  8
#define HB_ 2048
#define TD_ 256

// ---------------------------------------------------------------------------
// Kernel 1: kh_diag[b,h,k] -- only the einsum diagonal of kh is ever used
// (8e7 FLOPs instead of 1.4e11 for the full kh tensor).
// ---------------------------------------------------------------------------
__global__ void lsh_khdiag(const float* __restrict__ keys,
                           const float* __restrict__ topo,
                           const float* __restrict__ hash_proj,
                           const float* __restrict__ bias,
                           float* __restrict__ khd) {
  int idx = blockIdx.x * blockDim.x + threadIdx.x;
  if (idx >= B_ * H_ * HB_) return;
  int k = idx % HB_;
  int h = (idx / HB_) % H_;
  int b = idx / (H_ * HB_);
  const float* krow = keys + ((size_t)b * SK_ + k) * D_;
  const float* trow = topo + ((size_t)b * SK_ + k) * TD_;
  const float* wcol = hash_proj + (size_t)h * D_ * HB_ + k;
  float acc = 0.f;
  for (int d = 0; d < D_; ++d)  acc  += krow[d] * wcol[(size_t)d * HB_];
  float acct = 0.f;
  for (int d = 0; d < TD_; ++d) acct += trow[d] * wcol[(size_t)d * HB_];
  float p = acc + 0.5f * acct + bias[h * HB_ + k];
  khd[idx] = (p > 0.f) ? 1.f : ((p < 0.f) ? -1.f : 0.f);
}

// ---------------------------------------------------------------------------
// Kernel 2: elementwise f32 -> bf16 (4-wide), one-time pre-conversion.
// ---------------------------------------------------------------------------
__global__ void lsh_cvt_bf16(const float* __restrict__ s,
                             __bf16* __restrict__ d, int n4) {
  int i = blockIdx.x * blockDim.x + threadIdx.x;
  if (i >= n4) return;
  float4 f = ((const float4*)s)[i];
  v4bf o = {(__bf16)f.x, (__bf16)f.y, (__bf16)f.z, (__bf16)f.w};
  ((v4bf*)d)[i] = o;
}

// ---------------------------------------------------------------------------
// A-operand fetch helper: per lane, two contiguous 8-element (16B) runs at
// d = dbase+16*hi and d = dbase+16*hi+8 (k = 16*hi + s, matching B).
// ---------------------------------------------------------------------------
__device__ __forceinline__ v16bf load_a16x32(const __bf16* row, int dbase,
                                             int hi) {
  v8bf lo = *(const v8bf*)(row + dbase + (hi << 4));
  v8bf hh = *(const v8bf*)(row + dbase + 8 + (hi << 4));
  return __builtin_shufflevector(lo, hh, 0, 1, 2, 3, 4, 5, 6, 7, 8, 9, 10, 11,
                                 12, 13, 14, 15);
}

// ---------------------------------------------------------------------------
// Kernel 3 (fast path): fused qh GEMM on pre-converted bf16 + diagonal
// product + head-mean + threshold.
// ---------------------------------------------------------------------------
__global__ __launch_bounds__(256)
void lsh_main_bf16(const __bf16* __restrict__ qbf,
                   const __bf16* __restrict__ tbf,
                   const __bf16* __restrict__ wbf,
                   const float* __restrict__ bias,
                   const float* __restrict__ khd,
                   float* __restrict__ out) {
  __shared__ __attribute__((aligned(32))) __bf16 lds_w[16 * D_];  // [col][d]

  const int tid  = threadIdx.x;
  const int lane = tid & 31;
  const int wave = tid >> 5;
  const int hi   = lane >> 4;
  const int m    = lane & 15;
  const int col  = lane & 15;

  const int k0 = blockIdx.x * 16;
  const int q0 = blockIdx.y * 128 + wave * 16;
  const int b  = blockIdx.z;

  const __bf16* qrow = qbf + ((size_t)b * SQ_ + q0 + m) * D_;
  const __bf16* trow = tbf + ((size_t)b * SQ_ + q0 + m) * TD_;

  v8f simacc = {};

  for (int h = 0; h < H_; ++h) {
    // Stage Wbf[h, d, k0:k0+16] -> lds_w[c][d] (transposed bf16 copy).
    const __bf16* wp = wbf + (size_t)h * D_ * HB_ + k0;
    for (int i = tid; i < 16 * D_; i += 256) {
      int c = i & 15;
      int d = i >> 4;
      lds_w[c * D_ + d] = wp[(size_t)d * HB_ + c];
    }
    // Prefetch next head's slice into cache while this head computes
    // (global_prefetch_b8 path).
    if (h + 1 < H_) {
      const __bf16* wn = wp + (size_t)D_ * HB_;
      for (int i = tid; i < D_; i += 256)
        __builtin_prefetch(wn + (size_t)i * HB_, 0, 3);
    }
    __syncthreads();

    v8f acc  = {};
    v8f acct = {};
    const __bf16* bptr = &lds_w[col * D_ + (hi << 4)];

    // Steps 0..7: query WMMA + topology WMMA (d < TD_ = 256), no branches.
#pragma unroll 4
    for (int ds = 0; ds < 8; ++ds) {
      const int dbase = ds << 5;
      v16bf bmat = *(const v16bf*)(bptr + dbase);
      v16bf amat = load_a16x32(qrow, dbase, hi);
      acc = __builtin_amdgcn_wmma_f32_16x16x32_bf16(
          false, amat, false, bmat, (short)0, acc, false, false);
      v16bf tmat = load_a16x32(trow, dbase, hi);
      acct = __builtin_amdgcn_wmma_f32_16x16x32_bf16(
          false, tmat, false, bmat, (short)0, acct, false, false);
    }
    // Steps 8..31: pure query WMMA -- loads + v_wmma only.
#pragma unroll 4
    for (int ds = 8; ds < 32; ++ds) {
      const int dbase = ds << 5;
      v16bf bmat = *(const v16bf*)(bptr + dbase);
      v16bf amat = load_a16x32(qrow, dbase, hi);
      acc = __builtin_amdgcn_wmma_f32_16x16x32_bf16(
          false, amat, false, bmat, (short)0, acc, false, false);
    }

    float bv = bias[h * HB_ + k0 + col];
    float kd = khd[((size_t)b * H_ + h) * HB_ + k0 + col];
#pragma unroll
    for (int r = 0; r < 8; ++r) {
      float p = acc[r] + 0.5f * acct[r] + bv;
      float s = (p > 0.f) ? 1.f : ((p < 0.f) ? -1.f : 0.f);
      simacc[r] += s * kd;
    }
    __syncthreads();
  }

  const float inv = 1.0f / (8.0f * 2048.0f);
#pragma unroll
  for (int r = 0; r < 8; ++r) {
    int q = q0 + r + (hi << 3);
    float sim = simacc[r] * inv;
    out[((size_t)b * SQ_ + q) * HB_ + k0 + col] = (sim > 0.3f) ? 1.0f : 0.0f;
  }
}

// ---------------------------------------------------------------------------
// Kernel 3 (fallback, f32 inputs with in-loop cvt) -- used if ws too small.
// ---------------------------------------------------------------------------
__global__ __launch_bounds__(256)
void lsh_main_f32(const float* __restrict__ queries,
                  const float* __restrict__ topo,
                  const float* __restrict__ hash_proj,
                  const float* __restrict__ bias,
                  const float* __restrict__ khd,
                  float* __restrict__ out) {
  __shared__ __attribute__((aligned(32))) __bf16 lds_w[16 * D_];

  const int tid  = threadIdx.x;
  const int lane = tid & 31;
  const int wave = tid >> 5;
  const int hi   = lane >> 4;
  const int m    = lane & 15;
  const int col  = lane & 15;

  const int k0 = blockIdx.x * 16;
  const int q0 = blockIdx.y * 128 + wave * 16;
  const int b  = blockIdx.z;

  const float* qrow = queries + ((size_t)b * SQ_ + q0 + m) * D_;
  const float* trow = topo    + ((size_t)b * SQ_ + q0 + m) * TD_;

  v8f simacc = {};

  for (int h = 0; h < H_; ++h) {
    const float* wp = hash_proj + (size_t)h * D_ * HB_ + k0;
    for (int i = tid; i < 16 * D_; i += 256) {
      int c = i & 15;
      int d = i >> 4;
      lds_w[c * D_ + d] = (__bf16)wp[(size_t)d * HB_ + c];
    }
    __syncthreads();

    v8f acc  = {};
    v8f acct = {};
    const __bf16* bptr = &lds_w[col * D_ + (hi << 4)];

    for (int ds = 0; ds < 32; ++ds) {
      const int dbase = ds << 5;
      v16bf bmat = *(const v16bf*)(bptr + dbase);
      v16bf amat;
#pragma unroll
      for (int j = 0; j < 8; ++j) {
        int dd = dbase + ((j & 3) << 1) + (hi << 4) + ((j >> 2) << 3);
        float2 f = *(const float2*)(qrow + dd);
        amat[2 * j]     = (__bf16)f.x;
        amat[2 * j + 1] = (__bf16)f.y;
      }
      acc = __builtin_amdgcn_wmma_f32_16x16x32_bf16(
          false, amat, false, bmat, (short)0, acc, false, false);

      if (ds < 8) {
        v16bf tmat;
#pragma unroll
        for (int j = 0; j < 8; ++j) {
          int dd = dbase + ((j & 3) << 1) + (hi << 4) + ((j >> 2) << 3);
          float2 f = *(const float2*)(trow + dd);
          tmat[2 * j]     = (__bf16)f.x;
          tmat[2 * j + 1] = (__bf16)f.y;
        }
        acct = __builtin_amdgcn_wmma_f32_16x16x32_bf16(
            false, tmat, false, bmat, (short)0, acct, false, false);
      }
    }

    float bv = bias[h * HB_ + k0 + col];
    float kd = khd[((size_t)b * H_ + h) * HB_ + k0 + col];
#pragma unroll
    for (int r = 0; r < 8; ++r) {
      float p = acc[r] + 0.5f * acct[r] + bv;
      float s = (p > 0.f) ? 1.f : ((p < 0.f) ? -1.f : 0.f);
      simacc[r] += s * kd;
    }
    __syncthreads();
  }

  const float inv = 1.0f / (8.0f * 2048.0f);
#pragma unroll
  for (int r = 0; r < 8; ++r) {
    int q = q0 + r + (hi << 3);
    float sim = simacc[r] * inv;
    out[((size_t)b * SQ_ + q) * HB_ + k0 + col] = (sim > 0.3f) ? 1.0f : 0.0f;
  }
}

extern "C" void kernel_launch(void* const* d_in, const int* in_sizes, int n_in,
                              void* d_out, int out_size, void* d_ws, size_t ws_size,
                              hipStream_t stream) {
  const float* queries = (const float*)d_in[0];
  const float* keys    = (const float*)d_in[1];
  const float* topo    = (const float*)d_in[2];
  const float* hproj   = (const float*)d_in[3];
  const float* bias    = (const float*)d_in[4];
  float* out = (float*)d_out;

  float* khd = (float*)d_ws;                       // 128 KB
  const size_t off_w = 131072ull;                  // Wbf: 32 MB
  const size_t off_q = off_w + 33554432ull;        // Qbf:  8 MB
  const size_t off_t = off_q + 8388608ull;         // Tbf:  2 MB
  const size_t need  = off_t + 2097152ull;         // ~42.2 MB total

  lsh_khdiag<<<(B_ * H_ * HB_ + 255) / 256, 256, 0, stream>>>(
      keys, topo, hproj, bias, khd);

  dim3 grid(HB_ / 16, SQ_ / 128, B_);

  if (ws_size >= need) {
    __bf16* wbf = (__bf16*)((char*)d_ws + off_w);
    __bf16* qbf = (__bf16*)((char*)d_ws + off_q);
    __bf16* tbf = (__bf16*)((char*)d_ws + off_t);
    const int nW4 = (H_ * D_ * HB_) / 4;
    const int nQ4 = (B_ * SQ_ * D_) / 4;
    const int nT4 = (B_ * SQ_ * TD_) / 4;
    lsh_cvt_bf16<<<(nW4 + 255) / 256, 256, 0, stream>>>(hproj, wbf, nW4);
    lsh_cvt_bf16<<<(nQ4 + 255) / 256, 256, 0, stream>>>(queries, qbf, nQ4);
    lsh_cvt_bf16<<<(nT4 + 255) / 256, 256, 0, stream>>>(topo, tbf, nT4);
    lsh_main_bf16<<<grid, 256, 0, stream>>>(qbf, tbf, wbf, bias, khd, out);
  } else {
    lsh_main_f32<<<grid, 256, 0, stream>>>(queries, topo, hproj, bias, khd, out);
  }
}
